// RNNBlock_81286551044243
// MI455X (gfx1250) — matrix operand order repaired and checked by reading, not compiled, not run on hardware
//
#include <hip/hip_runtime.h>
#include <hip/hip_bf16.h>

// ---------------------------------------------------------------------------
// Problem dims
// ---------------------------------------------------------------------------
#define B_   64
#define T_   1024
#define IN_  256
#define H_   512
#define G4_  2048          // 4*H
#define NWG_REC 32         // workgroups in recurrent kernel (grid-synced)

typedef __bf16 bf16;
typedef __attribute__((ext_vector_type(8)))  bf16  v8bf;
typedef __attribute__((ext_vector_type(16))) bf16  v16bf;
typedef __attribute__((ext_vector_type(8)))  float v8f;

// ---------------------------------------------------------------------------
// Workspace layout (bytes)
// ---------------------------------------------------------------------------
static constexpr size_t OFF_XPROJ = 0;                                      // f32 [T][4 btile][128 ctile][32 lane][8]
static constexpr size_t SZ_XPROJ  = (size_t)T_ * B_ * G4_ * 4;              // 512 MB
static constexpr size_t OFF_XBF   = OFF_XPROJ + SZ_XPROJ;                   // bf16 x, t-major [T][B][IN]
static constexpr size_t SZ_XBF    = (size_t)T_ * B_ * IN_ * 2;              // 32 MB
static constexpr size_t OFF_WIH   = OFF_XBF + SZ_XBF;                       // bf16 [4H][IN]
static constexpr size_t SZ_WIH    = (size_t)G4_ * IN_ * 2;
static constexpr size_t OFF_CONV  = OFF_WIH + SZ_WIH;                       // bf16 [H][IN]
static constexpr size_t SZ_CONV   = (size_t)H_ * IN_ * 2;
static constexpr size_t OFF_HB0   = OFF_CONV + SZ_CONV;                     // bf16 [B][H] ping
static constexpr size_t SZ_HB     = (size_t)B_ * H_ * 2;
static constexpr size_t OFF_HB1   = OFF_HB0 + SZ_HB;                        // bf16 [B][H] pong
static constexpr size_t OFF_BAR   = OFF_HB1 + SZ_HB;                        // 2 x u32 barrier
static constexpr size_t WS_NEEDED = OFF_BAR + 256;

// ---------------------------------------------------------------------------
// Helpers
// ---------------------------------------------------------------------------
__device__ __forceinline__ v16bf cat16(v8bf lo, v8bf hi) {
  return __builtin_shufflevector(lo, hi, 0, 1, 2, 3, 4, 5, 6, 7, 8, 9, 10, 11, 12, 13, 14, 15);
}
__device__ __forceinline__ v8f wmma_bf16(v16bf a, v16bf b, v8f c) {
  return __builtin_amdgcn_wmma_f32_16x16x32_bf16(false, a, false, b, (short)0, c, false, false);
}
__device__ __forceinline__ float sigm_f(float x)  { return 1.0f / (1.0f + __expf(-x)); }
__device__ __forceinline__ float tanh_f(float x)  { return 1.0f - 2.0f / (1.0f + __expf(2.0f * x)); }

// A-operand (16x32 bf16): lane holds row m=lane&15; 16 contiguous K at
// kbase + (lane>>4)*16.
__device__ __forceinline__ v16bf load_a(const bf16* row, int kbase, int half) {
  const bf16* p = row + kbase + half * 16;
  return cat16(*(const v8bf*)p, *(const v8bf*)(p + 8));
}
// B-operand (32x16 bf16): lane holds col n=lane&15; 16 contiguous K at
// kbase + (lane>>4)*16 along the (transposed) weight row.
__device__ __forceinline__ v16bf load_b(const bf16* wrow, int kbase, int half) {
  const bf16* p = wrow + kbase + half * 16;
  return cat16(*(const v8bf*)p, *(const v8bf*)(p + 8));
}

// Grid-wide barrier over NWG_REC workgroups (atomic arrive + generation).
__device__ __forceinline__ void grid_barrier(unsigned* cnt, unsigned* gen, unsigned nwg) {
  __syncthreads();
  if (threadIdx.x == 0) {
    unsigned g = __hip_atomic_load(gen, __ATOMIC_RELAXED, __HIP_MEMORY_SCOPE_AGENT);
    if (__hip_atomic_fetch_add(cnt, 1u, __ATOMIC_ACQ_REL, __HIP_MEMORY_SCOPE_AGENT) == nwg - 1u) {
      __hip_atomic_store(cnt, 0u, __ATOMIC_RELAXED, __HIP_MEMORY_SCOPE_AGENT);
      __hip_atomic_store(gen, g + 1u, __ATOMIC_RELEASE, __HIP_MEMORY_SCOPE_AGENT);
    } else {
      while (__hip_atomic_load(gen, __ATOMIC_ACQUIRE, __HIP_MEMORY_SCOPE_AGENT) == g) {
        __builtin_amdgcn_s_sleep(8);
      }
    }
  }
  __syncthreads();
  __threadfence();
}

// ---------------------------------------------------------------------------
// K0: prep — bf16 casts, x transpose to t-major, h0 cast, barrier reset
// ---------------------------------------------------------------------------
__global__ void k0_prep(const float* __restrict__ x, const float* __restrict__ W_ih,
                        const float* __restrict__ conv_w, const float* __restrict__ h0,
                        bf16* __restrict__ xbf, bf16* __restrict__ wihbf,
                        bf16* __restrict__ convbf, bf16* __restrict__ hbuf0,
                        unsigned* __restrict__ bar) {
  int i = blockIdx.x * blockDim.x + threadIdx.x;
  if (i < B_ * T_ * IN_) {                       // x: [B,T,IN] f32 -> [T,B,IN] bf16
    int b = i / (T_ * IN_);
    int rem = i - b * (T_ * IN_);
    int t = rem / IN_, k = rem - t * IN_;
    xbf[((size_t)t * B_ + b) * IN_ + k] = (bf16)x[i];
  }
  if (i < G4_ * IN_) wihbf[i]  = (bf16)W_ih[i];
  if (i < H_ * IN_)  convbf[i] = (bf16)conv_w[i];
  if (i < B_ * H_)   hbuf0[i]  = (bf16)h0[i];
  if (i < 2)         bar[i]    = 0u;
}

// ---------------------------------------------------------------------------
// K1: x_proj = x @ W_ih.T + (b_ih + b_hh), written in packed WMMA-tile layout
//   row r = t*64 + b ; tile (t, btile, ctile): [32 lane][8 f32] contiguous
// ---------------------------------------------------------------------------
__global__ void __launch_bounds__(128) k1_xproj(const bf16* __restrict__ xbf,
                                                const bf16* __restrict__ wih,
                                                const float* __restrict__ b_ih,
                                                const float* __restrict__ b_hh,
                                                float* __restrict__ xproj) {
  const int lane = threadIdx.x & 31;
  const int wave = threadIdx.x >> 5;
  const int half = lane >> 4;
  const int n16  = lane & 15;
  const int rt   = blockIdx.x >> 3;                       // 0..4095 row tiles
  const int cg   = ((blockIdx.x & 7) << 2) + wave;        // 0..31 col groups (4 tiles each)

  v8f acc[4];
#pragma unroll
  for (int q = 0; q < 4; ++q)
#pragma unroll
    for (int r = 0; r < 8; ++r) acc[q][r] = 0.0f;

  const bf16* arow = xbf + (size_t)(rt * 16 + n16) * IN_;
#pragma unroll
  for (int kc = 0; kc < IN_ / 32; ++kc) {
    v16bf A = load_a(arow, kc * 32, half);
#pragma unroll
    for (int q = 0; q < 4; ++q) {
      int g = (cg * 4 + q) * 16 + n16;
      v16bf Bm = load_b(wih + (size_t)g * IN_, kc * 32, half);
      acc[q] = wmma_bf16(A, Bm, acc[q]);
    }
  }

  const int t = rt >> 2, bt = rt & 3;
#pragma unroll
  for (int q = 0; q < 4; ++q) {
    int ct = cg * 4 + q;
    float bias = b_ih[ct * 16 + n16] + b_hh[ct * 16 + n16];
#pragma unroll
    for (int r = 0; r < 8; ++r) acc[q][r] += bias;
    float* dst = xproj + ((((size_t)t * 4 + bt) * 128 + ct) * 32 + lane) * 8;
    *(v8f*)dst = acc[q];
  }
}

// ---------------------------------------------------------------------------
// K2: recurrence. 32 WGs x 128 thr. WG owns 16 hidden cols (+4 gate tiles).
//   W_hh slice pre-swizzled into LDS (B-operand layout); c in VGPRs; h in
//   global bf16 ping-pong; one grid barrier per timestep.
// ---------------------------------------------------------------------------
__global__ void __launch_bounds__(128) k2_lstm(const float* __restrict__ xproj,
                                               const float* __restrict__ W_hh,
                                               const float* __restrict__ c0,
                                               bf16* __restrict__ hbuf0,
                                               bf16* __restrict__ hbuf1,
                                               float* __restrict__ out,
                                               unsigned* __restrict__ bar) {
  __shared__ bf16 Bsm[64 * 512];                         // 64 KB: 64 tiles x 1KB
  const int wg   = blockIdx.x;                           // 0..31
  const int lane = threadIdx.x & 31;
  const int wave = threadIdx.x >> 5;                     // batch tile 0..3
  const int half = lane >> 4;
  const int n16  = lane & 15;

  // Pre-swizzle W_hh slice into LDS in exact B-operand layout.
  for (int e = threadIdx.x; e < 64 * 512; e += 128) {
    int q  = e >> 13;                 // gate 0..3
    int n  = (e >> 9) & 15;           // col within tile
    int k  = e & 511;
    int kc = k >> 5, ko = k & 31;
    int L  = n + ((ko >> 4) << 4);
    int j  = ko & 15;
    float w = W_hh[(size_t)(q * 512 + wg * 16 + n) * H_ + k];
    Bsm[(q * 16 + kc) * 512 + L * 16 + j] = (bf16)w;
  }
  __syncthreads();

  const int ncol = wg * 16 + n16;                        // hidden column
  v8f creg, hlast = {};
#pragma unroll
  for (int r = 0; r < 8; ++r) {
    int m = wave * 16 + r + half * 8;
    creg[r] = c0[m * H_ + ncol];
  }

  bf16* hb[2] = { hbuf0, hbuf1 };

  for (int t = 0; t < T_; ++t) {
    const bf16* hin  = hb[t & 1];
    bf16*       hout = hb[(t + 1) & 1];

    // C operand = x_proj tiles (bias already folded in)
    v8f acc[4];
#pragma unroll
    for (int q = 0; q < 4; ++q) {
      int ct = wg + 32 * q;
      acc[q] = *(const v8f*)(xproj + ((((size_t)t * 4 + wave) * 128 + ct) * 32 + lane) * 8);
    }
    if (t + 1 < T_) {                                    // hide next-step x_proj latency
#pragma unroll
      for (int q = 0; q < 4; ++q) {
        int ct = wg + 32 * q;
        __builtin_prefetch(xproj + ((((size_t)(t + 1) * 4 + wave) * 128 + ct) * 32 + lane) * 8, 0, 3);
      }
    }

    // gates += h @ W_hh.T  (K = 512, 16 k-chunks, 4 WMMAs each)
    const bf16* arow = hin + (size_t)(wave * 16 + n16) * H_;
#pragma unroll
    for (int kc = 0; kc < H_ / 32; ++kc) {
      v16bf A = load_a(arow, kc * 32, half);
#pragma unroll
      for (int q = 0; q < 4; ++q) {
        const bf16* bp = &Bsm[(q * 16 + kc) * 512 + lane * 16];
        v16bf Bm = cat16(*(const v8bf*)bp, *(const v8bf*)(bp + 8));
        acc[q] = wmma_bf16(A, Bm, acc[q]);
      }
    }

    // i,f,g,o share (m,n) layout across the 4 accumulators
#pragma unroll
    for (int r = 0; r < 8; ++r) {
      float iv = sigm_f(acc[0][r]);
      float fv = sigm_f(acc[1][r]);
      float gv = tanh_f(acc[2][r]);
      float ov = sigm_f(acc[3][r]);
      float cv = fv * creg[r] + iv * gv;
      creg[r]  = cv;
      float hv = ov * tanh_f(cv);
      hlast[r] = hv;
      int m = wave * 16 + r + half * 8;
      out[((size_t)m * T_ + t) * H_ + ncol] = hv;         // raw y (pre-LN)
      hout[m * H_ + ncol] = (bf16)hv;
    }

    grid_barrier(bar, bar + 1, NWG_REC);
  }

  // hT, cT
#pragma unroll
  for (int r = 0; r < 8; ++r) {
    int m = wave * 16 + r + half * 8;
    out[(size_t)B_ * T_ * H_ + m * H_ + ncol]           = hlast[r];
    out[(size_t)B_ * T_ * H_ + B_ * H_ + m * H_ + ncol] = creg[r];
  }
}

// ---------------------------------------------------------------------------
// K3: fused LayerNorm (unbiased std, /(std+eps)) + 1x1-conv skip (WMMA) + add
//   block = 16 rows (fixed b, t-range) x all 512 cols, 4 waves x 8 col tiles
// ---------------------------------------------------------------------------
__global__ void __launch_bounds__(128) k3_lnskip(float* __restrict__ out,
                                                 const bf16* __restrict__ xbf,
                                                 const bf16* __restrict__ convbf,
                                                 const float* __restrict__ gamma,
                                                 const float* __restrict__ beta,
                                                 const float* __restrict__ conv_b) {
  __shared__ float Ysm[16 * 512];
  __shared__ float Psum[16 * 8], Psq[16 * 8];
  __shared__ float Smean[16], Srd[16];
  const int b     = blockIdx.x >> 6;            // 0..63
  const int tbase = (blockIdx.x & 63) * 16;
  const int tid   = threadIdx.x;
  const int lane  = tid & 31, wave = tid >> 5, half = lane >> 4, n16 = lane & 15;

  float* yrow0 = out + ((size_t)b * T_ + tbase) * H_;
  for (int i = tid; i < 16 * 512; i += 128) Ysm[i] = yrow0[i];
  __syncthreads();

  {
    int row = tid >> 3, seg = tid & 7;
    float s = 0.f, ss = 0.f;
    const float* yr = &Ysm[row * 512 + seg * 64];
#pragma unroll
    for (int j = 0; j < 64; ++j) { float v = yr[j]; s += v; ss += v * v; }
    Psum[row * 8 + seg] = s; Psq[row * 8 + seg] = ss;
  }
  __syncthreads();
  if (tid < 16) {
    float s = 0.f, ss = 0.f;
#pragma unroll
    for (int j = 0; j < 8; ++j) { s += Psum[tid * 8 + j]; ss += Psq[tid * 8 + j]; }
    float mean = s * (1.0f / 512.0f);
    float var  = ss - 512.0f * mean * mean;
    var = var > 0.f ? var : 0.f;
    float stdv = __fsqrt_rn(var * (1.0f / 511.0f));     // unbiased (H-1)
    Smean[tid] = mean;
    Srd[tid]   = 1.0f / (stdv + 1e-5f);
  }
  __syncthreads();

  // skip = x @ conv_w.T ; A rows are 16 timesteps at fixed b (x is t-major)
  v8f acc[8];
#pragma unroll
  for (int q = 0; q < 8; ++q)
#pragma unroll
    for (int r = 0; r < 8; ++r) acc[q][r] = 0.0f;

  const bf16* arow = xbf + ((size_t)(tbase + n16) * B_ + b) * IN_;
#pragma unroll
  for (int kc = 0; kc < IN_ / 32; ++kc) {
    v16bf A = load_a(arow, kc * 32, half);
#pragma unroll
    for (int q = 0; q < 8; ++q) {
      int n = (wave * 8 + q) * 16 + n16;
      v16bf Bm = load_b(convbf + (size_t)n * IN_, kc * 32, half);
      acc[q] = wmma_bf16(A, Bm, acc[q]);
    }
  }

#pragma unroll
  for (int q = 0; q < 8; ++q) {
    int n = (wave * 8 + q) * 16 + n16;
    float gn = gamma[n], bn = beta[n], cn = conv_b[n];
#pragma unroll
    for (int r = 0; r < 8; ++r) {
      int m = r + half * 8;                              // local row -> t = tbase+m
      float yv = Ysm[m * 512 + n];
      float o  = gn * (yv - Smean[m]) * Srd[m] + bn + acc[q][r] + cn;
      out[((size_t)b * T_ + tbase + m) * H_ + n] = o;
    }
  }
}

// ---------------------------------------------------------------------------
// Launch
// ---------------------------------------------------------------------------
extern "C" void kernel_launch(void* const* d_in, const int* in_sizes, int n_in,
                              void* d_out, int out_size, void* d_ws, size_t ws_size,
                              hipStream_t stream) {
  (void)in_sizes; (void)n_in; (void)out_size;
  if (ws_size < WS_NEEDED) return;

  const float* x      = (const float*)d_in[0];
  const float* h0     = (const float*)d_in[1];
  const float* c0     = (const float*)d_in[2];
  const float* W_ih   = (const float*)d_in[3];
  const float* W_hh   = (const float*)d_in[4];
  const float* b_ih   = (const float*)d_in[5];
  const float* b_hh   = (const float*)d_in[6];
  const float* gamma  = (const float*)d_in[7];
  const float* beta   = (const float*)d_in[8];
  const float* conv_w = (const float*)d_in[9];
  const float* conv_b = (const float*)d_in[10];
  float* out = (float*)d_out;

  char* ws = (char*)d_ws;
  float*    xproj  = (float*)(ws + OFF_XPROJ);
  bf16*     xbf    = (bf16*)(ws + OFF_XBF);
  bf16*     wihbf  = (bf16*)(ws + OFF_WIH);
  bf16*     convbf = (bf16*)(ws + OFF_CONV);
  bf16*     hbuf0  = (bf16*)(ws + OFF_HB0);
  bf16*     hbuf1  = (bf16*)(ws + OFF_HB1);
  unsigned* bar    = (unsigned*)(ws + OFF_BAR);

  k0_prep<<<(B_ * T_ * IN_) / 256, 256, 0, stream>>>(x, W_ih, conv_w, h0,
                                                     xbf, wihbf, convbf, hbuf0, bar);
  k1_xproj<<<4096 * 8, 128, 0, stream>>>(xbf, wihbf, b_ih, b_hh, xproj);
  k2_lstm<<<NWG_REC, 128, 0, stream>>>(xproj, W_hh, c0, hbuf0, hbuf1, out, bar);
  k3_lnskip<<<B_ * (T_ / 16), 128, 0, stream>>>(out, xbf, convbf, gamma, beta, conv_b);
}
